// DeepFM_5050881540344
// MI455X (gfx1250) — matrix operand order repaired and checked by reading, not compiled, not run on hardware
//
#include <hip/hip_runtime.h>

typedef __attribute__((ext_vector_type(16))) __bf16 v16bf;
typedef __attribute__((ext_vector_type(8)))  __bf16 v8bf;
typedef __attribute__((ext_vector_type(8)))  float  v8f;

#define BATCH   16384
#define NFIELDS 26
#define EDIM    32
#define DIN     (NFIELDS * EDIM)   // 832
#define H1DIM   256
#define H2DIM   128
#define VOCAB   100000
#define BN_EPS  1e-5f

// ---------------------------------------------------------------------------
// 1) Weight convert + transpose: W (K,N) f32 row-major -> WT (N,K) bf16
// ---------------------------------------------------------------------------
__global__ __launch_bounds__(256) void conv_transpose_bf16(
    const float* __restrict__ W, __bf16* __restrict__ WT, int K, int N) {
  int i = blockIdx.x * blockDim.x + threadIdx.x;
  if (i >= K * N) return;
  int k = i / N, n = i - k * N;
  WT[(size_t)n * K + k] = (__bf16)W[i];
}

// ---------------------------------------------------------------------------
// 2) Embedding gather + FM (one wave32 per sample; lane == embedding dim)
// ---------------------------------------------------------------------------
__global__ __launch_bounds__(256) void embed_fm_kernel(
    const int* __restrict__ xc,          // (B, 26)
    const float* __restrict__ lin,       // (26, V, 1)
    const float* __restrict__ lat,       // (26, V, 32)
    const float* __restrict__ bias,      // (1,)
    __bf16* __restrict__ Xb,             // (B, 832) bf16
    float* __restrict__ fm)              // (B,)
{
  int wave = (blockIdx.x * blockDim.x + threadIdx.x) >> 5;
  int lane = threadIdx.x & 31;
  if (wave >= BATCH) return;
  const int b = wave;

  float s = 0.f, sumsq = 0.f;
#pragma unroll 1
  for (int f = 0; f < NFIELDS; ++f) {
    int idx = xc[b * NFIELDS + f];
    float v = lat[((size_t)f * VOCAB + idx) * EDIM + lane];  // coalesced 128B
    s += v;
    sumsq += v * v;
    Xb[(size_t)b * DIN + f * EDIM + lane] = (__bf16)v;
  }
  float linv = 0.f;
  if (lane < NFIELDS) {
    int idx = xc[b * NFIELDS + lane];
    linv = lin[(size_t)lane * VOCAB + idx];
  }
  float sq = s * s;            // square-of-sum contribution for dim=lane
#pragma unroll
  for (int m = 16; m >= 1; m >>= 1) {
    sq    += __shfl_xor(sq, m, 32);
    sumsq += __shfl_xor(sumsq, m, 32);
    linv  += __shfl_xor(linv, m, 32);
  }
  if (lane == 0) fm[b] = linv + 0.5f * (sq - sumsq) + bias[0];
}

// ---------------------------------------------------------------------------
// 3) WMMA bf16 GEMM: C(M,N) f32 = A(M,K) bf16 * Bt(N,K) bf16 (B transposed)
//    256 threads = 8 waves (4x2); each wave owns a 32x32 tile built from
//    2 A-frags x 2 B-frags -> 4 independent WMMA accumulator chains and
//    2 b128 loads per v_wmma (A/B fragments each reused twice).
//    Block macro-tile = 128 rows x 64 cols.
// ---------------------------------------------------------------------------
__global__ __launch_bounds__(256) void wmma_gemm_bf16(
    const __bf16* __restrict__ A, const __bf16* __restrict__ Bt,
    float* __restrict__ C, int M, int N, int K)
{
  const int lane   = threadIdx.x & 31;
  const int waveId = threadIdx.x >> 5;
  const int wr = waveId & 3;        // 4 wave-rows
  const int wc = waveId >> 2;       // 2 wave-cols
  const int tileRow = blockIdx.y * 128 + wr * 32;
  const int tileCol = blockIdx.x * 64 + wc * 32;

  const int l15 = lane & 15;
  const int h   = lane >> 4;        // K-half selector (k 16h..16h+15, contiguous order)

  const __bf16* arow0 = A  + (size_t)(tileRow + l15) * K;
  const __bf16* arow1 = arow0 + (size_t)16 * K;
  const __bf16* brow0 = Bt + (size_t)(tileCol + l15) * K;
  const __bf16* brow1 = brow0 + (size_t)16 * K;

  v8f acc00 = {}, acc01 = {}, acc10 = {}, acc11 = {};

  for (int k0 = 0; k0 < K; k0 += 32) {
    // A frag (16-bit A 16x32): lanes 0-15 hold K 0-15, lanes 16-31
    // hold K 16-31 -> one contiguous 32B load at k0+h*16 (matches B).
    v16bf af0 = *(const v16bf*)(arow0 + k0 + h * 16);
    v16bf af1 = *(const v16bf*)(arow1 + k0 + h * 16);

    // B frag (16-bit B 32x16): lane = column, lanes 0-15 K 0-15, lanes 16-31
    // K 16-31 -> one contiguous 32B row read of the transposed weights.
    v16bf bf0 = *(const v16bf*)(brow0 + k0 + h * 16);
    v16bf bf1 = *(const v16bf*)(brow1 + k0 + h * 16);

    acc00 = __builtin_amdgcn_wmma_f32_16x16x32_bf16(
        false, af0, false, bf0, (short)0, acc00, false, false);
    acc01 = __builtin_amdgcn_wmma_f32_16x16x32_bf16(
        false, af0, false, bf1, (short)0, acc01, false, false);
    acc10 = __builtin_amdgcn_wmma_f32_16x16x32_bf16(
        false, af1, false, bf0, (short)0, acc10, false, false);
    acc11 = __builtin_amdgcn_wmma_f32_16x16x32_bf16(
        false, af1, false, bf1, (short)0, acc11, false, false);
  }

  // C/D layout: VGPR i -> M = i (lanes 0-15) / i+8 (lanes 16-31), N = lane&15
#pragma unroll
  for (int i = 0; i < 8; ++i) {
    int r0 = tileRow + i + 8 * h;
    int r1 = r0 + 16;
    int c0 = tileCol + l15;
    int c1 = c0 + 16;
    C[(size_t)r0 * N + c0] = acc00[i];
    C[(size_t)r0 * N + c1] = acc01[i];
    C[(size_t)r1 * N + c0] = acc10[i];
    C[(size_t)r1 * N + c1] = acc11[i];
  }
}

// ---------------------------------------------------------------------------
// 4) BatchNorm: coalesced per-column partial sums, then finalize to affine
// ---------------------------------------------------------------------------
__global__ void bn_stats_partial(const float* __restrict__ Y, int rows, int cols,
                                 float* __restrict__ sumb, float* __restrict__ sqb) {
  int c = threadIdx.x;                       // blockDim.x == cols
  int rowsPer = rows / gridDim.x;
  int r0 = blockIdx.x * rowsPer;
  float sum = 0.f, sq = 0.f;
  for (int r = r0; r < r0 + rowsPer; ++r) {  // whole-block row reads coalesced
    float v = Y[(size_t)r * cols + c];
    sum += v; sq += v * v;
  }
  atomicAdd(&sumb[c], sum);
  atomicAdd(&sqb[c], sq);
}

__global__ void bn_finalize(const float* __restrict__ sumb, const float* __restrict__ sqb,
                            const float* __restrict__ g, const float* __restrict__ be,
                            float* __restrict__ Ac, float* __restrict__ Bc, int rows) {
  int c = threadIdx.x;                       // blockDim.x == cols
  float m   = sumb[c] / (float)rows;
  float var = sqb[c] / (float)rows - m * m;  // biased variance, as reference
  float a   = g[c] * rsqrtf(var + BN_EPS);
  Ac[c] = a;
  Bc[c] = be[c] - m * a;
}

__global__ __launch_bounds__(256) void bn_relu_bf16(
    const float* __restrict__ Y, const float* __restrict__ Ac,
    const float* __restrict__ Bc, __bf16* __restrict__ H, int total, int colsMask) {
  int i = blockIdx.x * blockDim.x + threadIdx.x;
  if (i >= total) return;
  int c = i & colsMask;
  float h = fmaf(Ac[c], Y[i], Bc[c]);
  H[i] = (__bf16)fmaxf(h, 0.f);
}

// ---------------------------------------------------------------------------
// 5) Fused BN2 + ReLU + dot(W3) + fm + b3 (one wave32 per row)
// ---------------------------------------------------------------------------
__global__ __launch_bounds__(256) void final_kernel(
    const float* __restrict__ Y2, const float* __restrict__ A2,
    const float* __restrict__ B2, const float* __restrict__ W3,
    const float* __restrict__ b3, const float* __restrict__ fm,
    float* __restrict__ out) {
  int wave = (blockIdx.x * blockDim.x + threadIdx.x) >> 5;
  int lane = threadIdx.x & 31;
  if (wave >= BATCH) return;
  float acc = 0.f;
#pragma unroll
  for (int j0 = 0; j0 < H2DIM; j0 += 32) {
    int j = j0 + lane;
    float h = fmaf(A2[j], Y2[(size_t)wave * H2DIM + j], B2[j]);
    acc = fmaf(fmaxf(h, 0.f), W3[j], acc);
  }
#pragma unroll
  for (int m = 16; m >= 1; m >>= 1) acc += __shfl_xor(acc, m, 32);
  if (lane == 0) out[wave] = fm[wave] + acc + b3[0];
}

// ---------------------------------------------------------------------------
// Launch
// ---------------------------------------------------------------------------
extern "C" void kernel_launch(void* const* d_in, const int* in_sizes, int n_in,
                              void* d_out, int out_size, void* d_ws, size_t ws_size,
                              hipStream_t stream) {
  const int*   xc   = (const int*)  d_in[0];
  const float* lin  = (const float*)d_in[1];
  const float* lat  = (const float*)d_in[2];
  const float* W1   = (const float*)d_in[3];
  // d_in[4] = b1 (cancels under BN), d_in[5]=g1, d_in[6]=be1
  const float* g1   = (const float*)d_in[5];
  const float* be1  = (const float*)d_in[6];
  const float* W2   = (const float*)d_in[7];
  // d_in[8] = b2 (cancels under BN)
  const float* g2   = (const float*)d_in[9];
  const float* be2  = (const float*)d_in[10];
  const float* W3   = (const float*)d_in[11];
  const float* b3   = (const float*)d_in[12];
  const float* bias = (const float*)d_in[13];
  float* out = (float*)d_out;

  char* ws = (char*)d_ws;
  // workspace layout (all offsets 256B aligned)
  const size_t XB_OFF  = 0;                                   // bf16 16384x832
  const size_t Y1_OFF  = XB_OFF  + (size_t)BATCH * DIN * 2;   // f32  16384x256
  const size_t H1_OFF  = Y1_OFF  + (size_t)BATCH * H1DIM * 4; // bf16 16384x256
  const size_t Y2_OFF  = H1_OFF  + (size_t)BATCH * H1DIM * 2; // f32  16384x128
  const size_t W1T_OFF = Y2_OFF  + (size_t)BATCH * H2DIM * 4; // bf16 256x832
  const size_t W2T_OFF = W1T_OFF + (size_t)H1DIM * DIN * 2;   // bf16 128x256
  const size_t FM_OFF  = W2T_OFF + (size_t)H2DIM * H1DIM * 2; // f32  16384
  const size_t ST_OFF  = FM_OFF  + (size_t)BATCH * 4;         // stats block

  __bf16* Xb  = (__bf16*)(ws + XB_OFF);
  float*  Y1  = (float*) (ws + Y1_OFF);
  __bf16* H1  = (__bf16*)(ws + H1_OFF);
  float*  Y2  = (float*) (ws + Y2_OFF);
  __bf16* W1T = (__bf16*)(ws + W1T_OFF);
  __bf16* W2T = (__bf16*)(ws + W2T_OFF);
  float*  fm  = (float*) (ws + FM_OFF);
  float*  sum1 = (float*)(ws + ST_OFF);            // 256
  float*  sq1  = sum1 + 256;                       // 256
  float*  A1   = sq1  + 256;                       // 256
  float*  B1   = A1   + 256;                       // 256
  float*  sum2 = B1   + 256;                       // 128
  float*  sq2  = sum2 + 128;                       // 128
  float*  A2   = sq2  + 128;                       // 128
  float*  B2   = A2   + 128;                       // 128

  // zero the stats accumulators (graph-capturable)
  hipMemsetAsync(sum1, 0, (256 + 256) * sizeof(float), stream);
  hipMemsetAsync(sum2, 0, (128 + 128) * sizeof(float), stream);

  // weight prep
  conv_transpose_bf16<<<(DIN * H1DIM + 255) / 256, 256, 0, stream>>>(W1, W1T, DIN, H1DIM);
  conv_transpose_bf16<<<(H1DIM * H2DIM + 255) / 256, 256, 0, stream>>>(W2, W2T, H1DIM, H2DIM);

  // embedding gather + FM (8 waves/block -> 2048 blocks)
  embed_fm_kernel<<<BATCH / 8, 256, 0, stream>>>(xc, lin, lat, bias, Xb, fm);

  // GEMM1: Y1 = Xb @ W1  (M=16384, N=256, K=832); block tile 128x64
  {
    dim3 grid(H1DIM / 64, BATCH / 128);
    wmma_gemm_bf16<<<grid, 256, 0, stream>>>(Xb, W1T, Y1, BATCH, H1DIM, DIN);
  }
  bn_stats_partial<<<64, H1DIM, 0, stream>>>(Y1, BATCH, H1DIM, sum1, sq1);
  bn_finalize<<<1, H1DIM, 0, stream>>>(sum1, sq1, g1, be1, A1, B1, BATCH);
  bn_relu_bf16<<<(BATCH * H1DIM + 255) / 256, 256, 0, stream>>>(
      Y1, A1, B1, H1, BATCH * H1DIM, H1DIM - 1);

  // GEMM2: Y2 = H1 @ W2  (M=16384, N=128, K=256); block tile 128x64
  {
    dim3 grid(H2DIM / 64, BATCH / 128);
    wmma_gemm_bf16<<<grid, 256, 0, stream>>>(H1, W2T, Y2, BATCH, H2DIM, H1DIM);
  }
  bn_stats_partial<<<64, H2DIM, 0, stream>>>(Y2, BATCH, H2DIM, sum2, sq2);
  bn_finalize<<<1, H2DIM, 0, stream>>>(sum2, sq2, g2, be2, A2, B2, BATCH);

  // fused BN2 + ReLU + W3 dot + fm + b3
  final_kernel<<<BATCH / 8, 256, 0, stream>>>(Y2, A2, B2, W3, b3, fm, out);
}
